// decoder_78752520340008
// MI455X (gfx1250) — matrix operand order repaired and checked by reading, not compiled, not run on hardware
//
#include <hip/hip_runtime.h>

typedef __attribute__((ext_vector_type(16))) __bf16 v16bf;
typedef __attribute__((ext_vector_type(8)))  __bf16 v8bf;
typedef __attribute__((ext_vector_type(8)))  float  v8f;

#define B_   64
#define TD_  512
#define TE_  1024
#define E_   512
#define U_   256
#define NC_  64
#define ZD_  1024   // 4*U
#define AK_  1024   // A = [x(256) | ctx(512) | h(256)]
#define NGRP 4
#define NWGG 16     // workgroups per group

// ---------------------------------------------------------------------------
// WMMA helpers (CDNA5 16x16x32 bf16, f32 accumulate)
// ---------------------------------------------------------------------------
__device__ inline v8f wmma_bf(v16bf a, v16bf b, v8f c) {
  return __builtin_amdgcn_wmma_f32_16x16x32_bf16(false, a, false, b, (short)0, c,
                                                 false, false);
}

// A-frag (16xK row-major), source f32 (global or LDS). ISA 7.12.2 layout:
// lanes 0-15: row=lane, halves hold K k0+{0..7}, k0+{16..23};
// lanes 16-31: row=lane-16, halves hold K k0+{8..15}, k0+{24..31}.
__device__ inline v16bf load_a_f32(const float* A, int lda, int row0, int k0, int lane) {
  const float* p = A + (long)(row0 + (lane & 15)) * lda + k0 + ((lane >> 4) << 3);
  v16bf a;
#pragma unroll
  for (int i = 0; i < 8; ++i) { a[i] = (__bf16)p[i]; a[8 + i] = (__bf16)p[16 + i]; }
  return a;
}

// A-frag from bf16 LDS panel: two contiguous 16B chunks -> ds_load_b128 x2
__device__ inline v16bf load_a_bf16v(const __bf16* A, int lda, int k0, int lane) {
  const __bf16* p = A + (lane & 15) * lda + k0 + ((lane >> 4) << 3);
  v8bf lo = *(const v8bf*)p;
  v8bf hi = *(const v8bf*)(p + 16);
  return __builtin_shufflevector(lo, hi, 0, 1, 2, 3, 4, 5, 6, 7,
                                 8, 9, 10, 11, 12, 13, 14, 15);
}

// B-frag from fragment-packed weights: one aligned 32B load per lane.
// Packed layout: tile=(k0/32)*tilesN+(col0/16); dst[(tile*32+lane)*16 + i]
__device__ inline v16bf load_b_pk(const __bf16* P, int tilesN, int k0, int col0, int lane) {
  const __bf16* p = P + ((((long)(k0 >> 5) * tilesN + (col0 >> 4)) << 5) + lane) * 16;
  return *(const v16bf*)p;
}

// ---------------------------------------------------------------------------
// Weight pre-pack: f32 [K,N] row-major -> bf16 WMMA B-fragment order.
// lane<16 holds rows k0+{0..7,16..23} of col col0+lane; lane>=16 rows k0+{8..15,24..31}.
// ---------------------------------------------------------------------------
__global__ void pack_b_k(const float* __restrict__ src, __bf16* __restrict__ dst,
                         int K, int N, int koff) {
  long idx = (long)blockIdx.x * blockDim.x + threadIdx.x;
  long tot = (long)K * N;
  long st  = (long)gridDim.x * blockDim.x;
  int tilesN = N >> 4;
  for (; idx < tot; idx += st) {
    int k = (int)(idx / N), n = (int)(idx % N);
    int kg = k + koff;
    int kr = kg & 31;
    long tile = (long)(kg >> 5) * tilesN + (n >> 4);
    int lane  = (((kr >> 3) & 1) << 4) + (n & 15);
    int i     = (kr & 7) + ((kr >> 4) << 3);
    dst[((tile << 5) + lane) * 16 + i] = (__bf16)src[idx];
  }
}

// ---------------------------------------------------------------------------
// Generic GEMM: C[M,N](f32 or bf16) = A[M,K](f32) @ Bpk[K,N](packed bf16)+bias
// ---------------------------------------------------------------------------
__global__ __launch_bounds__(256)
void gemm_bf16_k(const float* __restrict__ A, const __bf16* __restrict__ Bpk,
                 const float* __restrict__ bias, float* __restrict__ Cf,
                 __bf16* __restrict__ Cb, int M, int N, int K) {
  int lane = threadIdx.x & 31;
  int wid  = blockIdx.x * 8 + (threadIdx.x >> 5);
  int tilesN = N >> 4;
  if (wid >= (M >> 4) * tilesN) return;
  int row0 = (wid / tilesN) << 4;
  int col0 = (wid % tilesN) << 4;
  v8f acc = {};
  for (int k0 = 0; k0 < K; k0 += 32)
    acc = wmma_bf(load_a_f32(A, K, row0, k0, lane),
                  load_b_pk(Bpk, tilesN, k0, col0, lane), acc);
  int c  = col0 + (lane & 15);
  int r0 = row0 + ((lane >> 4) << 3);
  float bv = bias[c];
  if (Cf) {
#pragma unroll
    for (int r = 0; r < 8; ++r) Cf[(long)(r0 + r) * N + c] = acc[r] + bv;
  } else {
#pragma unroll
    for (int r = 0; r < 8; ++r) Cb[(long)(r0 + r) * N + c] = (__bf16)(acc[r] + bv);
  }
}

// ---------------------------------------------------------------------------
// Utility kernels
// ---------------------------------------------------------------------------
__global__ void cvt_bf16_k(const float* __restrict__ s, __bf16* __restrict__ d, long n) {
  long i  = (long)blockIdx.x * blockDim.x + threadIdx.x;
  long st = (long)gridDim.x * blockDim.x;
  for (; i < n; i += st) d[i] = (__bf16)s[i];
}

__global__ __launch_bounds__(256)
void rownorm256_k(const float* __restrict__ x, float* __restrict__ y, int rows) {
  int lane = threadIdx.x & 31;
  int row  = blockIdx.x * 8 + (threadIdx.x >> 5);
  if (row >= rows) return;
  const float* p = x + (long)row * 256;
  float v[8]; float s = 0.f;
#pragma unroll
  for (int i = 0; i < 8; ++i) { v[i] = p[lane + (i << 5)]; s += v[i]; }
  for (int m = 16; m; m >>= 1) s += __shfl_xor(s, m, 32);
  float mean = s * (1.f / 256.f);
  float q = 0.f;
#pragma unroll
  for (int i = 0; i < 8; ++i) { float d = v[i] - mean; q += d * d; }
  for (int m = 16; m; m >>= 1) q += __shfl_xor(q, m, 32);
  float rs = rsqrtf(q * (1.f / 256.f) + 1e-4f);
  float* o = y + (long)row * 256;
#pragma unroll
  for (int i = 0; i < 8; ++i) o[lane + (i << 5)] = (v[i] - mean) * rs;
}

__global__ __launch_bounds__(256)
void softmax64_k(const float* __restrict__ x, float* __restrict__ y, int rows) {
  int lane = threadIdx.x & 31;
  int row  = blockIdx.x * 8 + (threadIdx.x >> 5);
  if (row >= rows) return;
  const float* p = x + (long)row * 64;
  float a0 = p[lane], a1 = p[lane + 32];
  float mx = fmaxf(a0, a1);
  for (int m = 16; m; m >>= 1) mx = fmaxf(mx, __shfl_xor(mx, m, 32));
  float e0 = __expf(a0 - mx), e1 = __expf(a1 - mx);
  float s = e0 + e1;
  for (int m = 16; m; m >>= 1) s += __shfl_xor(s, m, 32);
  float inv = 1.f / s;
  float* q = y + (long)row * 64;
  q[lane] = e0 * inv; q[lane + 32] = e1 * inv;
}

// ---------------------------------------------------------------------------
// Group barrier: CDNA5 cluster barrier when clustered, else device-atomic
// monotonic barrier over the 16 WGs of a batch-group.
// ---------------------------------------------------------------------------
__device__ inline void group_barrier(unsigned* ctr) {
  __threadfence();
  __syncthreads();
  if (threadIdx.x == 0) {
    if (__builtin_amdgcn_cluster_id_x() != 0) {
      __builtin_amdgcn_s_cluster_barrier();   // s_barrier_signal/-wait -3
    } else {
      unsigned old    = atomicAdd(ctr, 1u);
      unsigned target = (old / (unsigned)NWGG + 1u) * (unsigned)NWGG;
      while (atomicAdd(ctr, 0u) < target) __builtin_amdgcn_s_sleep(1);
    }
  }
  __syncthreads();
  __threadfence();
}

__device__ inline float sigm(float x) { return 1.f / (1.f + __expf(-x)); }

// ---------------------------------------------------------------------------
// Recurrent attention-LSTM layer. grid = 64 WGs (4 groups x 16), block = 256.
// ---------------------------------------------------------------------------
__global__ __launch_bounds__(256)
void attend_k(const float* __restrict__ xseq,      // [B,TD,U] f32
              float* __restrict__ yseq,            // [B,TD,U] f32
              const __bf16* __restrict__ enc,      // [B,TE,U] bf16
              const __bf16* __restrict__ speech,   // [B,TE,E] bf16
              const __bf16* __restrict__ Wa,       // packed [U,U]
              const float* __restrict__ Wa_b,      // [U]
              const float* __restrict__ va,        // [U]
              const float* __restrict__ va_b,      // [1]
              const __bf16* __restrict__ Wz,       // packed [AK,ZD] rows K|R
              const float* __restrict__ bz,        // [ZD]
              float* __restrict__ s_buf,           // [NGRP,16,U]
              float* __restrict__ logit_buf,       // [NGRP,16,TE]
              __bf16* __restrict__ A_buf,          // [NGRP,16,AK]
              float* __restrict__ z_buf,           // [NGRP,16,ZD]
              unsigned* __restrict__ bar)          // [NGRP*16]
{
  const int g    = blockIdx.x >> 4;
  const int wg   = blockIdx.x & 15;
  const int b0   = g << 4;
  const int tid  = threadIdx.x;
  const int lane = tid & 31;
  const int wav  = tid >> 5;

  float*    sg = s_buf     + (long)g * 16 * U_;
  float*    lg = logit_buf + (long)g * 16 * TE_;
  __bf16*   Ag = A_buf     + (long)g * 16 * AK_;
  float*    zg = z_buf     + (long)g * 16 * ZD_;
  unsigned* bg = bar + g * 16;

  __shared__ float  h_lds[16][U_];                    // 16 KB
  __shared__ float  c_lds[16][U_];                    // 16 KB
  __shared__ __bf16 w_lds[16][TE_];                   // 32 KB softmax weights
  __shared__ __align__(32) __bf16 a_lds[16][AK_];     // 32 KB A panel stage

  for (int i = tid; i < 16 * U_; i += 256) {
    (&h_lds[0][0])[i] = 0.f; (&c_lds[0][0])[i] = 0.f;
  }
  __syncthreads();

  for (int t = 0; t < TD_; ++t) {
    // ---- phase A: s = h @ Wa + Wa_b ; cols [16wg, 16wg+16)
    if (wav == 0) {
      const int col0 = wg << 4;
      v8f acc = {};
      for (int k0 = 0; k0 < U_; k0 += 32)
        acc = wmma_bf(load_a_f32(&h_lds[0][0], U_, 0, k0, lane),
                      load_b_pk(Wa, U_ >> 4, k0, col0, lane), acc);
      int c  = col0 + (lane & 15);
      int r0 = (lane >> 4) << 3;
      float bv = Wa_b[c];
#pragma unroll
      for (int r = 0; r < 8; ++r) sg[(r0 + r) * U_ + c] = acc[r] + bv;
    }
    group_barrier(bg);                                           // B1

    // ---- phase B: logits[b,t'] for this WG's Te slice (64 rows)
    {
      const float vab = va_b[0];
      for (int p = wav; p < 16 * 64; p += 8) {
        const int b  = p >> 6;
        const int tg = (wg << 6) + (p & 63);
        const __bf16* ep = enc + ((long)(b0 + b) * TE_ + tg) * U_;
        const float*  sp = sg + b * U_;
        const int ub = lane << 3;
        float acc = 0.f;
#pragma unroll
        for (int i = 0; i < 8; ++i)
          acc += tanhf(sp[ub + i] + (float)ep[ub + i]) * va[ub + i];
        for (int m = 16; m; m >>= 1) acc += __shfl_xor(acc, m, 32);
        if (lane == 0) lg[b * TE_ + tg] = acc + vab;
      }
    }
    group_barrier(bg);                                           // B2

    // ---- phase C: softmax over Te -> w_lds (redundant per WG)
    for (int b = wav; b < 16; b += 8) {
      const float* lp = lg + b * TE_;
      float loc[32], mx = -3.4e38f;
#pragma unroll
      for (int i = 0; i < 32; ++i) { loc[i] = lp[lane + (i << 5)]; mx = fmaxf(mx, loc[i]); }
      for (int m = 16; m; m >>= 1) mx = fmaxf(mx, __shfl_xor(mx, m, 32));
      float sum = 0.f;
#pragma unroll
      for (int i = 0; i < 32; ++i) { loc[i] = __expf(loc[i] - mx); sum += loc[i]; }
      for (int m = 16; m; m >>= 1) sum += __shfl_xor(sum, m, 32);
      float inv = 1.f / sum;
#pragma unroll
      for (int i = 0; i < 32; ++i) w_lds[b][lane + (i << 5)] = (__bf16)(loc[i] * inv);
    }
    __syncthreads();

    // ---- phase D: ctx e-slice (32 cols), assemble A = [x | ctx | h]
    {
      const int e = (wg << 5) + lane;
      for (int bi = 0; bi < 2; ++bi) {
        const int b = (wav << 1) + bi;
        const __bf16* sp = speech + (long)(b0 + b) * TE_ * E_ + e;
        float acc = 0.f;
        for (int tt = 0; tt < TE_; ++tt) {
          if ((tt & 63) == 0)
            __builtin_prefetch(sp + (long)(tt + 64) * E_, 0, 1);  // global_prefetch
          acc += (float)w_lds[b][tt] * (float)sp[(long)tt * E_];
        }
        Ag[b * AK_ + U_ + e] = (__bf16)acc;
      }
    }
    if (wg == 0) {
      for (int i = tid; i < 16 * U_; i += 256) {
        const int b = i >> 8, u = i & 255;
        Ag[b * AK_ + u]          = (__bf16)xseq[((long)(b0 + b) * TD_ + t) * U_ + u];
        Ag[b * AK_ + 3 * U_ + u] = (__bf16)h_lds[b][u];
      }
    }
    group_barrier(bg);                                           // B3

    // ---- phase E: async-stage A panel to LDS, then z = A @ Wz + bz
    {
      // 16x1024 bf16 = 32KB staged with CDNA5 async global->LDS B128 copies
      unsigned lbase = (unsigned)(size_t)(&a_lds[0][0]);
      for (int i = tid; i < (16 * AK_) / 8; i += 256) {      // 16B chunks
        unsigned    lo = lbase + (unsigned)i * 16u;
        const char* ga = (const char*)Ag + (long)i * 16;
        asm volatile("global_load_async_to_lds_b128 %0, %1, off"
                     :: "v"(lo), "v"(ga) : "memory");
      }
      asm volatile("s_wait_asynccnt 0x0" ::: "memory");
      __syncthreads();
    }
    if (wav < 4) {
      const int col0 = (wg << 6) + (wav << 4);
      v8f acc = {};
      for (int k0 = 0; k0 < AK_; k0 += 32)
        acc = wmma_bf(load_a_bf16v(&a_lds[0][0], AK_, k0, lane),
                      load_b_pk(Wz, ZD_ >> 4, k0, col0, lane), acc);
      int c  = col0 + (lane & 15);
      int r0 = (lane >> 4) << 3;
      float bv = bz[c];
#pragma unroll
      for (int r = 0; r < 8; ++r) zg[(r0 + r) * ZD_ + c] = acc[r] + bv;
    }
    group_barrier(bg);                                           // B4

    // ---- phase F: LSTM gates (i,f,g,o), redundant per WG
    for (int i = tid; i < 16 * U_; i += 256) {
      const int b = i >> 8, u = i & 255;
      const float* zp = zg + b * ZD_;
      float ig = sigm(zp[u]);
      float fg = sigm(zp[U_ + u]);
      float gg = tanhf(zp[2 * U_ + u]);
      float og = sigm(zp[3 * U_ + u]);
      float cn = fg * c_lds[b][u] + ig * gg;
      float hn = og * tanhf(cn);
      c_lds[b][u] = cn;
      h_lds[b][u] = hn;
      if (wg == 0) yseq[((long)(b0 + b) * TD_ + t) * U_ + u] = hn;
    }
    group_barrier(bg);                                           // B5
  }
}

// ---------------------------------------------------------------------------
// Host launcher
// ---------------------------------------------------------------------------
extern "C" void kernel_launch(void* const* d_in, const int* in_sizes, int n_in,
                              void* d_out, int out_size, void* d_ws, size_t ws_size,
                              hipStream_t stream) {
  (void)in_sizes; (void)n_in; (void)out_size; (void)ws_size;
  const float* trans  = (const float*)d_in[0];
  const float* speech = (const float*)d_in[1];
  const float* emb_W  = (const float*)d_in[2];
  const float* emb_b  = (const float*)d_in[3];
  const float* Ua_W[2] = {(const float*)d_in[4],  (const float*)d_in[13]};
  const float* Ua_b[2] = {(const float*)d_in[5],  (const float*)d_in[14]};
  const float* Wa_W[2] = {(const float*)d_in[6],  (const float*)d_in[15]};
  const float* Wa_b[2] = {(const float*)d_in[7],  (const float*)d_in[16]};
  const float* va_W[2] = {(const float*)d_in[8],  (const float*)d_in[17]};
  const float* va_b[2] = {(const float*)d_in[9],  (const float*)d_in[18]};
  const float* Km[2]   = {(const float*)d_in[10], (const float*)d_in[19]};
  const float* Rm[2]   = {(const float*)d_in[11], (const float*)d_in[20]};
  const float* bz[2]   = {(const float*)d_in[12], (const float*)d_in[21]};
  const float* m1_W = (const float*)d_in[22];
  const float* m1_b = (const float*)d_in[23];
  const float* m2_W = (const float*)d_in[24];
  const float* m2_b = (const float*)d_in[25];
  const float* d_W  = (const float*)d_in[26];
  const float* d_b  = (const float*)d_in[27];
  float* out = (float*)d_out;

  char* ws = (char*)d_ws;
  size_t off = 0;
  auto alloc = [&](size_t bytes) -> char* {
    off = (off + 255) & ~(size_t)255;
    char* p = ws + off; off += bytes; return p;
  };

  __bf16* speech_bf = (__bf16*)alloc((size_t)B_ * TE_ * E_ * 2);
  __bf16* enc_bf    = (__bf16*)alloc((size_t)B_ * TE_ * U_ * 2);
  float*  xseq      = (float*)alloc((size_t)B_ * TD_ * U_ * 4);
  float*  yseq      = (float*)alloc((size_t)B_ * TD_ * U_ * 4);
  float*  zfin      = (float*)alloc((size_t)B_ * TD_ * NC_ * 4);
  __bf16* emb_pk    = (__bf16*)alloc((size_t)NC_ * U_ * 2);
  __bf16* Ua_pk[2]  = {(__bf16*)alloc((size_t)E_ * U_ * 2), (__bf16*)alloc((size_t)E_ * U_ * 2)};
  __bf16* Wa_pk[2]  = {(__bf16*)alloc((size_t)U_ * U_ * 2), (__bf16*)alloc((size_t)U_ * U_ * 2)};
  __bf16* Wz_pk[2]  = {(__bf16*)alloc((size_t)AK_ * ZD_ * 2), (__bf16*)alloc((size_t)AK_ * ZD_ * 2)};
  __bf16* m1_pk     = (__bf16*)alloc((size_t)U_ * U_ * 2);
  __bf16* m2_pk     = (__bf16*)alloc((size_t)U_ * U_ * 2);
  __bf16* d_pk      = (__bf16*)alloc((size_t)U_ * NC_ * 2);
  float*  s_buf     = (float*)alloc((size_t)NGRP * 16 * U_ * 4);
  float*  l_buf     = (float*)alloc((size_t)NGRP * 16 * TE_ * 4);
  __bf16* A_buf     = (__bf16*)alloc((size_t)NGRP * 16 * AK_ * 2);
  float*  z_buf     = (float*)alloc((size_t)NGRP * 16 * ZD_ * 4);
  unsigned* bar     = (unsigned*)alloc((size_t)NGRP * 16 * 4);

  auto pack = [&](const float* s, __bf16* d, int K, int N, int koff) {
    long tot = (long)K * N;
    pack_b_k<<<(int)((tot + 255) / 256), 256, 0, stream>>>(s, d, K, N, koff);
  };
  auto gemm = [&](const float* A, const __bf16* Bpk, const float* bias,
                  float* Cf, __bf16* Cb, int M, int N, int K) {
    int tiles = (M / 16) * (N / 16);
    gemm_bf16_k<<<(tiles + 7) / 8, 256, 0, stream>>>(A, Bpk, bias, Cf, Cb, M, N, K);
  };

  // conversions + weight fragment-packing
  {
    long n = (long)B_ * TE_ * E_;
    cvt_bf16_k<<<(int)((n + 255) / 256), 256, 0, stream>>>(speech, speech_bf, n);
  }
  pack(emb_W, emb_pk, NC_, U_, 0);
  for (int L = 0; L < 2; ++L) {
    pack(Ua_W[L], Ua_pk[L], E_, U_, 0);
    pack(Wa_W[L], Wa_pk[L], U_, U_, 0);
    pack(Km[L], Wz_pk[L], U_ + E_, ZD_, 0);        // K rows 0..767
    pack(Rm[L], Wz_pk[L], U_, ZD_, U_ + E_);       // R rows 768..1023
  }
  pack(m1_W, m1_pk, U_, U_, 0);
  pack(m2_W, m2_pk, U_, U_, 0);
  pack(d_W,  d_pk,  U_, NC_, 0);

  const int rows = B_ * TD_;   // 32768

  // secrets = trans @ emb_W + emb_b
  gemm(trans, emb_pk, emb_b, xseq, nullptr, rows, U_, NC_);

  for (int L = 0; L < 2; ++L) {
    // enc = speech @ Ua + Ua_b (bf16, L2-resident)
    gemm(speech, Ua_pk[L], Ua_b[L], nullptr, enc_bf, B_ * TE_, U_, E_);
    attend_k<<<NGRP * NWGG, 256, 0, stream>>>(
        xseq, yseq, enc_bf, speech_bf, Wa_pk[L], Wa_b[L], va_W[L], va_b[L],
        Wz_pk[L], bz[L], s_buf, l_buf, A_buf, z_buf, bar);
    rownorm256_k<<<rows / 8, 256, 0, stream>>>(yseq, xseq, rows);
  }

  gemm(xseq, m1_pk, m1_b, yseq, nullptr, rows, U_, U_);
  rownorm256_k<<<rows / 8, 256, 0, stream>>>(yseq, xseq, rows);
  gemm(xseq, m2_pk, m2_b, yseq, nullptr, rows, U_, U_);
  rownorm256_k<<<rows / 8, 256, 0, stream>>>(yseq, xseq, rows);
  gemm(xseq, d_pk, d_b, zfin, nullptr, rows, NC_, U_);
  softmax64_k<<<rows / 8, 256, 0, stream>>>(zfin, out, rows);
}
